// BlockwiseDense_8735963480154
// MI455X (gfx1250) — compile-verified, matrix operand order checked by reading, not run on hardware
//
#include <hip/hip_runtime.h>

typedef _Float16 v4h  __attribute__((ext_vector_type(4)));
typedef _Float16 v8h  __attribute__((ext_vector_type(8)));
typedef _Float16 v16h __attribute__((ext_vector_type(16)));
typedef float    v8f  __attribute__((ext_vector_type(8)));

#define NUM_ROWS 256   // l per block (K of the GEMM)
#define NUM_COLS 256   // k per block (N of the GEMM)
#define IN_SIZE  4096
#define BATCH    128
#define KCHUNK   64    // l-chunk staged in LDS
#define MB       64    // batch rows per workgroup (half of 128)

// Nearest of 256 log-spaced RRAM conductance levels:
//   levels[n] = g_min + scale*(1 - exp(-tau*n/255)), scale=(g_inf-g_min)/(1-exp(-tau))
// Analytic inversion to index space, then compare the two bracketing levels
// (mirrors the reference's searchsorted + clip(idx,1,255) lo/hi pick; relu
// zeros land on levels[0] = g_min, values above the top snap to g_max).
__device__ __forceinline__ float rram_q(float c) {
    const float tau   = 0.75f;
    const float g_min = 0.001f;
    const float scale = (2.0f - 0.001f) / (1.0f - __expf(-tau));
    float w = fmaxf(c, 0.0f);
    float t = (w - g_min) * (1.0f / scale);
    t = fminf(fmaxf(t, 0.0f), 0.9999999f);
    float nf = (-255.0f / tau) * __logf(1.0f - t);
    nf = fminf(nf, 255.0f);
    float n1 = fminf(floorf(nf) + 1.0f, 255.0f);
    float n0 = n1 - 1.0f;
    float lo = g_min + scale * (1.0f - __expf(n0 * (-tau / 255.0f)));
    float hi = g_min + scale * (1.0f - __expf(n1 * (-tau / 255.0f)));
    return ((w - lo) < (hi - w)) ? lo : hi;
}

// Load one 16(N)x64(K) B row-pair for column-tile base pointer p:
// two K=32 fragments laid out per the ISA f16 B layout (lane groups at +kb).
__device__ __forceinline__ void load_brow(const _Float16* p, v16h& b0, v16h& b1) {
    v8h l0 = *(const v8h*)(p);
    v8h h0 = *(const v8h*)(p + 16);
    v8h l1 = *(const v8h*)(p + 32);
    v8h h1 = *(const v8h*)(p + 48);
    b0 = __builtin_shufflevector(l0, h0, 0,1,2,3,4,5,6,7,8,9,10,11,12,13,14,15);
    b1 = __builtin_shufflevector(l1, h1, 0,1,2,3,4,5,6,7,8,9,10,11,12,13,14,15);
}

__global__ __launch_bounds__(128)
void blockwise_dense_rram(const float* __restrict__ x,
                          const float* __restrict__ cores,
                          float* __restrict__ y) {
    const int j  = blockIdx.x;   // input block
    const int i  = blockIdx.y;   // output block
    const int bh = blockIdx.z;   // batch half

    __shared__ _Float16 lds_w[NUM_COLS * KCHUNK]; // [k][lc], lc contiguous (B operand, K-major)
    __shared__ _Float16 lds_x[MB       * KCHUNK]; // [b][lc], lc contiguous (A operand, K-major)

    const int tid  = threadIdx.x;
    const int lane = tid & 31;
    const int wave = tid >> 5;

    const float* Wg = cores + (size_t)(i * 16 + j) * (NUM_COLS * NUM_ROWS); // [k][l]
    const float* Xg = x + (size_t)(bh * MB) * IN_SIZE + (size_t)j * NUM_ROWS;

    v8f acc[16] = {};

    const int nlo = lane & 15;
    const int kb  = (lane < 16) ? 0 : 8;          // ISA f16 A/B lane-group K offset
    const int mrow = wave * 16 + nlo;             // A row this lane feeds

    for (int kc = 0; kc < NUM_ROWS; kc += KCHUNK) {
        // ---- stage quantized W chunk: 256 x 64 f32 -> f16 LDS (coalesced float4) ----
        #pragma unroll 4
        for (int t = 0; t < 32; ++t) {
            int f  = tid + 128 * t;               // 0..4095 float4 slots
            int k  = f >> 4;
            int lc = (f & 15) << 2;
            const float4 v = *(const float4*)(Wg + (size_t)k * NUM_ROWS + kc + lc);
            v4h h;
            h[0] = (_Float16)rram_q(v.x);
            h[1] = (_Float16)rram_q(v.y);
            h[2] = (_Float16)rram_q(v.z);
            h[3] = (_Float16)rram_q(v.w);
            *(v4h*)(lds_w + k * KCHUNK + lc) = h;
        }
        // ---- stage x chunk: 64 x 64 f32 -> f16 LDS ----
        #pragma unroll 4
        for (int t = 0; t < 8; ++t) {
            int f  = tid + 128 * t;               // 0..1023 float4 slots
            int b  = f >> 4;
            int lc = (f & 15) << 2;
            const float4 v = *(const float4*)(Xg + (size_t)b * IN_SIZE + kc + lc);
            v4h h;
            h[0] = (_Float16)v.x;
            h[1] = (_Float16)v.y;
            h[2] = (_Float16)v.z;
            h[3] = (_Float16)v.w;
            *(v4h*)(lds_x + b * KCHUNK + lc) = h;
        }
        __syncthreads();

        // ---- A fragments for this chunk (invariant over all 16 N tiles) ----
        const _Float16* xrow = lds_x + mrow * KCHUNK + kb;
        v8h a0l = *(const v8h*)(xrow);
        v8h a0h = *(const v8h*)(xrow + 16);
        v8h a1l = *(const v8h*)(xrow + 32);
        v8h a1h = *(const v8h*)(xrow + 48);
        v16h a0 = __builtin_shufflevector(a0l, a0h, 0,1,2,3,4,5,6,7,8,9,10,11,12,13,14,15);
        v16h a1 = __builtin_shufflevector(a1l, a1h, 0,1,2,3,4,5,6,7,8,9,10,11,12,13,14,15);

        // ---- software-pipelined B rows: prefetch nt+1 while WMMAs run on nt ----
        const _Float16* wbase = lds_w + nlo * KCHUNK + kb;
        v16h b0, b1, nb0, nb1;
        load_brow(wbase, b0, b1);
        #pragma unroll
        for (int nt = 0; nt < 16; ++nt) {
            if (nt < 15)
                load_brow(wbase + (nt + 1) * 16 * KCHUNK, nb0, nb1);
            acc[nt] = __builtin_amdgcn_wmma_f32_16x16x32_f16(
                false, a0, false, b0, (short)0, acc[nt], false, false);
            acc[nt] = __builtin_amdgcn_wmma_f32_16x16x32_f16(
                false, a1, false, b1, (short)0, acc[nt], false, false);
            b0 = nb0;
            b1 = nb1;
        }
        __syncthreads();
    }

    // ---- store D: VGPR r holds M=r (lanes 0-15) / M=r+8 (lanes 16-31), N=lane&15 ----
    #pragma unroll
    for (int nt = 0; nt < 16; ++nt) {
        #pragma unroll
        for (int r = 0; r < 8; ++r) {
            int m = wave * 16 + ((lane < 16) ? r : r + 8);
            int b = bh * MB + m;
            int k = nt * 16 + nlo;
            y[(((size_t)b * 16 + i) * 16 + j) * NUM_COLS + k] = acc[nt][r];
        }
    }
}

extern "C" void kernel_launch(void* const* d_in, const int* in_sizes, int n_in,
                              void* d_out, int out_size, void* d_ws, size_t ws_size,
                              hipStream_t stream) {
    const float* x     = (const float*)d_in[0];  // (128, 4096) f32
    const float* cores = (const float*)d_in[1];  // (16, 16, 256, 256) f32
    float* y           = (float*)d_out;          // (128, 16, 16, 256) f32
    (void)in_sizes; (void)n_in; (void)out_size; (void)d_ws; (void)ws_size;

    dim3 grid(16, 16, 2);   // (j, i, batch-half)
    dim3 block(128);        // 4 waves of 32
    blockwise_dense_rram<<<grid, block, 0, stream>>>(x, cores, y);
}
